// TTS_Simple_25623774888619
// MI455X (gfx1250) — matrix operand order repaired and checked by reading, not compiled, not run on hardware
//
#include <hip/hip_runtime.h>
#include <hip/hip_bf16.h>

typedef __bf16 bf16;
typedef __attribute__((ext_vector_type(16))) __bf16 v16bf;
typedef __attribute__((ext_vector_type(8)))  float  v8f;

union BFrag { v16bf v; float4 f[2]; };

// ---- WMMA fragment helpers (layouts per CDNA5 ISA 7.12.2, wave32) ----

// A (16x32 bf16): lane holds row (lane&15); K elems {khalf*8+0..7, 16+khalf*8+0..7}
__device__ inline v16bf load_a_frag(const bf16* row, int kb) {
  int khalf = (threadIdx.x >> 4) & 1;
  const bf16* p = row + kb + khalf * 8;
  BFrag fr;
  fr.f[0] = *reinterpret_cast<const float4*>(p);
  fr.f[1] = *reinterpret_cast<const float4*>(p + 16);
  return fr.v;
}
// B (32x16 bf16): lane holds column (lane&15); K elems {khalf*8+0..7, 16+khalf*8+0..7}
__device__ inline v16bf load_b_frag(const bf16* row, int kb) {
  int khalf = (threadIdx.x >> 4) & 1;
  const bf16* p = row + kb + khalf * 8;
  BFrag fr;
  fr.f[0] = *reinterpret_cast<const float4*>(p);
  fr.f[1] = *reinterpret_cast<const float4*>(p + 16);
  return fr.v;
}
__device__ inline v16bf zero_frag() {
  BFrag fr;
  fr.f[0] = make_float4(0.f, 0.f, 0.f, 0.f);
  fr.f[1] = make_float4(0.f, 0.f, 0.f, 0.f);
  return fr.v;
}
__device__ inline v8f zero_acc() {
  v8f a;
  for (int j = 0; j < 8; ++j) a[j] = 0.f;
  return a;
}
__device__ inline v8f wmma_bf16(v16bf a, v16bf b, v8f c) {
  return __builtin_amdgcn_wmma_f32_16x16x32_bf16(false, a, false, b, (short)0, c,
                                                 false, false);
}
// C/D: lane element j -> row m0 + 8*khalf + j, col n
__device__ inline void store_acc(float* C, int ldc, int m0, int n, v8f acc) {
  int khalf = (threadIdx.x >> 4) & 1;
  for (int j = 0; j < 8; ++j)
    C[(size_t)(m0 + khalf * 8 + j) * ldc + n] = acc[j];
}

__device__ inline float sigmf(float x) { return 1.f / (1.f + expf(-x)); }

// ---------------- prep kernels ----------------

__global__ void k_cvt_bf16(const float* __restrict__ s, bf16* __restrict__ d, int n) {
  int i = blockIdx.x * blockDim.x + threadIdx.x;
  if (i < n) d[i] = (bf16)s[i];
}
__global__ void k_add2(const float* __restrict__ a, const float* __restrict__ b,
                       float* __restrict__ o, int n) {
  int i = blockIdx.x * blockDim.x + threadIdx.x;
  if (i < n) o[i] = a[i] + b[i];
}
// conv weight (Eo=512, Ei=512, 3) -> wB[n][k*512+e] = w[n][e][k]  (K contiguous per n)
__global__ void k_convw(const float* __restrict__ w, bf16* __restrict__ o) {
  int i = blockIdx.x * blockDim.x + threadIdx.x;
  if (i >= 512 * 1536) return;
  int n = i / 1536, r = i % 1536, k = r / 512, e = r % 512;
  o[i] = (bf16)w[(size_t)n * 1536 + e * 3 + k];
}
__global__ void k_embed(const int* __restrict__ x, const float* __restrict__ emb,
                        bf16* __restrict__ xe) {
  int i = blockIdx.x * blockDim.x + threadIdx.x;  // over 8192*512
  if (i >= 8192 * 512) return;
  int row = i >> 9, e = i & 511;
  xe[i] = (bf16)emb[(size_t)x[row] * 512 + e];
}

// ---------------- conv (3-tap) as WMMA GEMM: (8192 x 512) = A(8192 x 1536)*B ----

__global__ void k_conv_gemm(const bf16* __restrict__ src, const bf16* __restrict__ wB,
                            const float* __restrict__ bias, float* __restrict__ out) {
  int wave = threadIdx.x >> 5, lane = threadIdx.x & 31;
  int r0 = blockIdx.x * 32;                          // 256 row blocks of 32
  int n  = blockIdx.y * 128 + wave * 16 + (lane & 15);
  const bf16* wrow = wB + (size_t)n * 1536;
  v8f acc0 = zero_acc(), acc1 = zero_acc();
  int ra = r0 + (lane & 15), rb = ra + 16;
  for (int tap = 0; tap < 3; ++tap) {
    int la = (ra & 255) + tap - 1;
    int lb = (rb & 255) + tap - 1;
    bool va = (la >= 0) && (la < 256);               // zero-pad + batch boundary
    bool vb = (lb >= 0) && (lb < 256);
    const bf16* arow0 = src + (size_t)(ra + tap - 1) * 512;
    const bf16* arow1 = src + (size_t)(rb + tap - 1) * 512;
    int kw0 = tap * 512;
    for (int kb = 0; kb < 512; kb += 32) {
      v16bf bfr = load_b_frag(wrow, kw0 + kb);
      v16bf a0 = va ? load_a_frag(arow0, kb) : zero_frag();
      v16bf a1 = vb ? load_a_frag(arow1, kb) : zero_frag();
      acc0 = wmma_bf16(a0, bfr, acc0);
      acc1 = wmma_bf16(a1, bfr, acc1);
    }
  }
  float bv = bias[n];
  int khalf = (lane >> 4) & 1;
  for (int j = 0; j < 8; ++j) {
    out[(size_t)(r0 + khalf * 8 + j) * 512 + n]      = acc0[j] + bv;
    out[(size_t)(r0 + 16 + khalf * 8 + j) * 512 + n] = acc1[j] + bv;
  }
}

// batch-norm (training-mode batch stats, biased var)
__global__ void k_bn_stats(const float* __restrict__ x, float* __restrict__ mean,
                           float* __restrict__ rstd) {
  int ch = blockIdx.x;  // 512 channels
  float s = 0.f, s2 = 0.f;
  for (int r = threadIdx.x; r < 8192; r += 256) {
    float v = x[(size_t)r * 512 + ch];
    s += v; s2 += v * v;
  }
  __shared__ float sh[256], sh2[256];
  sh[threadIdx.x] = s; sh2[threadIdx.x] = s2;
  __syncthreads();
  for (int st = 128; st > 0; st >>= 1) {
    if (threadIdx.x < st) { sh[threadIdx.x] += sh[threadIdx.x + st];
                            sh2[threadIdx.x] += sh2[threadIdx.x + st]; }
    __syncthreads();
  }
  if (threadIdx.x == 0) {
    float mu = sh[0] / 8192.f;
    float var = sh2[0] / 8192.f - mu * mu;
    mean[ch] = mu;
    rstd[ch] = rsqrtf(var + 1e-5f);
  }
}
__global__ void k_bn_apply(const float* __restrict__ x, const float* __restrict__ mean,
                           const float* __restrict__ rstd, const float* __restrict__ g,
                           const float* __restrict__ beta, bf16* __restrict__ o) {
  int i = blockIdx.x * blockDim.x + threadIdx.x;
  if (i >= 8192 * 512) return;
  int ch = i & 511;
  float v = (x[i] - mean[ch]) * rstd[ch] * g[ch] + beta[ch];
  o[i] = (bf16)fmaxf(v, 0.f);
}

// ---------------- encoder biLSTM ----------------

__global__ void k_enc_zero(float* hF, float* cF, float* hB, float* cB,
                           bf16* hbF, bf16* hbB) {
  int i = blockIdx.x * blockDim.x + threadIdx.x;
  if (i >= 32 * 512) return;
  hF[i] = 0.f; cF[i] = 0.f; hB[i] = 0.f; cB[i] = 0.f;
  hbF[i] = (bf16)0.f; hbB[i] = (bf16)0.f;
}

// gates = h @ whh^T + x_t @ wih^T ; grid (16 Nblocks, 2 dirs), 256 thr (8 waves x 16 cols)
__global__ void k_enc_gates(const bf16* __restrict__ henc,
                            const bf16* __restrict__ hbF, const bf16* __restrict__ hbB,
                            const bf16* __restrict__ whhF, const bf16* __restrict__ wihF,
                            const bf16* __restrict__ whhB, const bf16* __restrict__ wihB,
                            const int* __restrict__ lens, int t,
                            float* __restrict__ gates) {
  int dir = blockIdx.y;
  const bf16* hb  = dir ? hbB  : hbF;
  const bf16* whh = dir ? whhB : whhF;
  const bf16* wih = dir ? wihB : wihF;
  int wave = threadIdx.x >> 5, lane = threadIdx.x & 31;
  int n = blockIdx.x * 128 + wave * 16 + (lane & 15);
  v8f acc0 = zero_acc(), acc1 = zero_acc();
  // recurrent term, K=512
  const bf16* wr1 = whh + (size_t)n * 512;
  const bf16* ar0 = hb + (size_t)(lane & 15) * 512;
  const bf16* ar1 = hb + (size_t)((lane & 15) + 16) * 512;
  for (int kb = 0; kb < 512; kb += 32) {
    v16bf bfr = load_b_frag(wr1, kb);
    acc0 = wmma_bf16(load_a_frag(ar0, kb), bfr, acc0);
    acc1 = wmma_bf16(load_a_frag(ar1, kb), bfr, acc1);
  }
  // input term, K=512 (backward dir gathers reversed row per batch)
  int b0 = lane & 15, b1 = b0 + 16;
  int t0 = t, t1 = t;
  if (dir) {
    int u0 = lens[b0] - 1 - t; t0 = u0 < 0 ? 0 : (u0 > 255 ? 255 : u0);
    int u1 = lens[b1] - 1 - t; t1 = u1 < 0 ? 0 : (u1 > 255 ? 255 : u1);
  }
  const bf16* xr0 = henc + ((size_t)b0 * 256 + t0) * 512;
  const bf16* xr1 = henc + ((size_t)b1 * 256 + t1) * 512;
  const bf16* wr2 = wih + (size_t)n * 512;
  for (int kb = 0; kb < 512; kb += 32) {
    v16bf bfr = load_b_frag(wr2, kb);
    acc0 = wmma_bf16(load_a_frag(xr0, kb), bfr, acc0);
    acc1 = wmma_bf16(load_a_frag(xr1, kb), bfr, acc1);
  }
  float* g = gates + (size_t)dir * 32 * 2048;
  store_acc(g, 2048, 0, n, acc0);
  store_acc(g, 2048, 16, n, acc1);
}

__global__ void k_enc_update(const float* __restrict__ gates,
                             const float* __restrict__ bF, const float* __restrict__ bB,
                             const int* __restrict__ lens, int t,
                             float* hF, float* cF, bf16* hbF,
                             float* hB, float* cB, bf16* hbB) {
  int dir = blockIdx.y;
  int i = blockIdx.x * blockDim.x + threadIdx.x;
  if (i >= 32 * 512) return;
  int b = i >> 9, j = i & 511;
  if (t >= lens[b]) return;  // packed-sequence freeze
  const float* g = gates + (size_t)dir * 32 * 2048 + (size_t)b * 2048;
  const float* bias = dir ? bB : bF;
  float* h = dir ? hB : hF; float* c = dir ? cB : cF; bf16* hb = dir ? hbB : hbF;
  float gi = g[j]        + bias[j];
  float gf = g[512 + j]  + bias[512 + j];
  float gg = g[1024 + j] + bias[1024 + j];
  float go = g[1536 + j] + bias[1536 + j];
  float cn = sigmf(gf) * c[i] + sigmf(gi) * tanhf(gg);
  float hn = sigmf(go) * tanhf(cn);
  c[i] = cn; h[i] = hn; hb[i] = (bf16)hn;
}

// ---------------- decoder ----------------

__global__ void k_dec_init(const float* hF, const float* hB, const float* cF,
                           const float* cB, float* hd, float* cd, bf16* hdb,
                           bf16* melb) {
  int i = blockIdx.x * blockDim.x + threadIdx.x;
  if (i < 32 * 1024) {
    int b = i >> 10, j = i & 1023;
    float hv = (j < 512) ? hF[b * 512 + j] : hB[b * 512 + (j - 512)];
    float cv = (j < 512) ? cF[b * 512 + j] : cB[b * 512 + (j - 512)];
    hd[i] = hv; cd[i] = cv; hdb[i] = (bf16)hv;
  }
  if (i < 32 * 128) melb[i] = (bf16)0.f;  // sos input
}

// gates(32 x 4096) = hd @ dwhh^T + mel @ dwih^T ; grid 32 blocks x 256 thr
__global__ void k_dec_gates(const bf16* __restrict__ hdb, const bf16* __restrict__ melb,
                            const bf16* __restrict__ dwhh, const bf16* __restrict__ dwih,
                            float* __restrict__ gates) {
  int wave = threadIdx.x >> 5, lane = threadIdx.x & 31;
  int n = blockIdx.x * 128 + wave * 16 + (lane & 15);
  v8f acc0 = zero_acc(), acc1 = zero_acc();
  const bf16* wr1 = dwhh + (size_t)n * 1024;
  const bf16* ar0 = hdb + (size_t)(lane & 15) * 1024;
  const bf16* ar1 = hdb + (size_t)((lane & 15) + 16) * 1024;
  for (int kb = 0; kb < 1024; kb += 32) {
    v16bf bfr = load_b_frag(wr1, kb);
    acc0 = wmma_bf16(load_a_frag(ar0, kb), bfr, acc0);
    acc1 = wmma_bf16(load_a_frag(ar1, kb), bfr, acc1);
  }
  const bf16* wr2 = dwih + (size_t)n * 128;
  const bf16* mr0 = melb + (size_t)(lane & 15) * 128;
  const bf16* mr1 = melb + (size_t)((lane & 15) + 16) * 128;
  for (int kb = 0; kb < 128; kb += 32) {
    v16bf bfr = load_b_frag(wr2, kb);
    acc0 = wmma_bf16(load_a_frag(mr0, kb), bfr, acc0);
    acc1 = wmma_bf16(load_a_frag(mr1, kb), bfr, acc1);
  }
  store_acc(gates, 4096, 0, n, acc0);
  store_acc(gates, 4096, 16, n, acc1);
}

// fused LSTM pointwise + mel projection (WMMA) + gate dot + masked output writes
__global__ void k_dec_update_proj(const float* __restrict__ gates,
                                  const float* __restrict__ bdec,
                                  float* __restrict__ hd, float* __restrict__ cd,
                                  bf16* __restrict__ hdb,
                                  const bf16* __restrict__ projw,
                                  const float* __restrict__ projb,
                                  const float* __restrict__ gatew,
                                  const float* __restrict__ gateb,
                                  const int* __restrict__ mlens, int t,
                                  bf16* __restrict__ melb,
                                  float* __restrict__ out_mel,
                                  float* __restrict__ out_gate,
                                  float* __restrict__ out_mask) {
  int tid = threadIdx.x;
  for (int i = tid; i < 32 * 1024; i += blockDim.x) {
    int b = i >> 10, j = i & 1023;
    const float* g = gates + (size_t)b * 4096;
    float gi = g[j]        + bdec[j];
    float gf = g[1024 + j] + bdec[1024 + j];
    float gg = g[2048 + j] + bdec[2048 + j];
    float go = g[3072 + j] + bdec[3072 + j];
    float cn = sigmf(gf) * cd[i] + sigmf(gi) * tanhf(gg);
    float hn = sigmf(go) * tanhf(cn);
    cd[i] = cn; hd[i] = hn; hdb[i] = (bf16)hn;
  }
  __threadfence();
  __syncthreads();
  // projection: 8 waves x 16 cols = 128 mel channels, M=32, K=1024
  int wave = tid >> 5, lane = tid & 31;
  int n = wave * 16 + (lane & 15);
  v8f acc0 = zero_acc(), acc1 = zero_acc();
  const bf16* wr = projw + (size_t)n * 1024;
  const bf16* ar0 = hdb + (size_t)(lane & 15) * 1024;
  const bf16* ar1 = hdb + (size_t)((lane & 15) + 16) * 1024;
  for (int kb = 0; kb < 1024; kb += 32) {
    v16bf bfr = load_b_frag(wr, kb);
    acc0 = wmma_bf16(load_a_frag(ar0, kb), bfr, acc0);
    acc1 = wmma_bf16(load_a_frag(ar1, kb), bfr, acc1);
  }
  float pb = projb[n];
  int khalf = (lane >> 4) & 1;
  for (int j = 0; j < 8; ++j) {
    int b0 = khalf * 8 + j;
    float m0 = acc0[j] + pb;
    melb[b0 * 128 + n] = (bf16)m0;  // feedback uses UNmasked mel
    out_mel[((size_t)b0 * 800 + t) * 128 + n] = (t > mlens[b0]) ? 0.f : m0;
    int b1 = 16 + khalf * 8 + j;
    float m1 = acc1[j] + pb;
    melb[b1 * 128 + n] = (bf16)m1;
    out_mel[((size_t)b1 * 800 + t) * 128 + n] = (t > mlens[b1]) ? 0.f : m1;
  }
  if (tid < 32) {  // stop-gate: 32 f32 dot products of length 1024
    float s = gateb[0];
    const float* hrow = hd + (size_t)tid * 1024;
    for (int j = 0; j < 1024; ++j) s += hrow[j] * gatew[j];
    bool msk = t > mlens[tid];
    out_gate[(size_t)tid * 800 + t] = msk ? 1000.f : s;
    out_mask[(size_t)tid * 800 + t] = msk ? 1.f : 0.f;
  }
}

// ---------------- host ----------------

extern "C" void kernel_launch(void* const* d_in, const int* in_sizes, int n_in,
                              void* d_out, int out_size, void* d_ws, size_t ws_size,
                              hipStream_t stream) {
  (void)in_sizes; (void)n_in; (void)out_size; (void)ws_size;
  const int*   x     = (const int*)d_in[0];
  const int*   tlens = (const int*)d_in[1];
  const int*   mlens = (const int*)d_in[3];   // d_in[2]=y unused
  const float* emb   = (const float*)d_in[4];
  const float* c1w = (const float*)d_in[5];  const float* c1b = (const float*)d_in[6];
  const float* bn1g = (const float*)d_in[7]; const float* bn1b = (const float*)d_in[8];
  const float* c2w = (const float*)d_in[9];  const float* c2b = (const float*)d_in[10];
  const float* bn2g = (const float*)d_in[11]; const float* bn2b = (const float*)d_in[12];
  const float* wihF = (const float*)d_in[13]; const float* whhF = (const float*)d_in[14];
  const float* bihF = (const float*)d_in[15]; const float* bhhF = (const float*)d_in[16];
  const float* wihB = (const float*)d_in[17]; const float* whhB = (const float*)d_in[18];
  const float* bihB = (const float*)d_in[19]; const float* bhhB = (const float*)d_in[20];
  const float* dwih = (const float*)d_in[21]; const float* dwhh = (const float*)d_in[22];
  const float* dbih = (const float*)d_in[23]; const float* dbhh = (const float*)d_in[24];
  const float* projw = (const float*)d_in[25]; const float* projb = (const float*)d_in[26];
  const float* gatew = (const float*)d_in[27]; const float* gateb = (const float*)d_in[28];

  char* w = (char*)d_ws;
  size_t off = 0;
  auto take = [&](size_t bytes) {
    char* p = w + off;
    off += (bytes + 255) & ~(size_t)255;
    return p;
  };
  bf16*  xe     = (bf16*) take((size_t)8192 * 512 * 2);
  float* cpre   = (float*)take((size_t)8192 * 512 * 4);
  bf16*  bn1o   = (bf16*) take((size_t)8192 * 512 * 2);
  bf16*  henc   = (bf16*) take((size_t)8192 * 512 * 2);
  bf16*  wB1    = (bf16*) take((size_t)512 * 1536 * 2);
  bf16*  wB2    = (bf16*) take((size_t)512 * 1536 * 2);
  bf16*  whhFb  = (bf16*) take((size_t)2048 * 512 * 2);
  bf16*  wihFb  = (bf16*) take((size_t)2048 * 512 * 2);
  bf16*  whhBb  = (bf16*) take((size_t)2048 * 512 * 2);
  bf16*  wihBb  = (bf16*) take((size_t)2048 * 512 * 2);
  bf16*  dwhhb  = (bf16*) take((size_t)4096 * 1024 * 2);
  bf16*  dwihb  = (bf16*) take((size_t)4096 * 128 * 2);
  bf16*  projwb = (bf16*) take((size_t)128 * 1024 * 2);
  float* bencF  = (float*)take(2048 * 4);
  float* bencB  = (float*)take(2048 * 4);
  float* bdec   = (float*)take(4096 * 4);
  float* mean   = (float*)take(512 * 4);
  float* rstd   = (float*)take(512 * 4);
  float* hF = (float*)take(32 * 512 * 4); float* cF = (float*)take(32 * 512 * 4);
  float* hB = (float*)take(32 * 512 * 4); float* cB = (float*)take(32 * 512 * 4);
  bf16*  hbF = (bf16*)take(32 * 512 * 2); bf16* hbB = (bf16*)take(32 * 512 * 2);
  float* egates = (float*)take((size_t)2 * 32 * 2048 * 4);
  float* hd = (float*)take(32 * 1024 * 4); float* cd = (float*)take(32 * 1024 * 4);
  bf16*  hdb  = (bf16*)take(32 * 1024 * 2);
  bf16*  melb = (bf16*)take(32 * 128 * 2);
  float* dgates = (float*)take((size_t)32 * 4096 * 4);

  auto cvt = [&](const float* s, bf16* d, int n) {
    k_cvt_bf16<<<(n + 255) / 256, 256, 0, stream>>>(s, d, n);
  };
  cvt(whhF, whhFb, 2048 * 512); cvt(wihF, wihFb, 2048 * 512);
  cvt(whhB, whhBb, 2048 * 512); cvt(wihB, wihBb, 2048 * 512);
  cvt(dwhh, dwhhb, 4096 * 1024); cvt(dwih, dwihb, 4096 * 128);
  cvt(projw, projwb, 128 * 1024);
  k_add2<<<8, 256, 0, stream>>>(bihF, bhhF, bencF, 2048);
  k_add2<<<8, 256, 0, stream>>>(bihB, bhhB, bencB, 2048);
  k_add2<<<16, 256, 0, stream>>>(dbih, dbhh, bdec, 4096);
  k_convw<<<(512 * 1536 + 255) / 256, 256, 0, stream>>>(c1w, wB1);
  k_convw<<<(512 * 1536 + 255) / 256, 256, 0, stream>>>(c2w, wB2);

  k_embed<<<(8192 * 512 + 255) / 256, 256, 0, stream>>>(x, emb, xe);
  dim3 cg(256, 4);
  k_conv_gemm<<<cg, 256, 0, stream>>>(xe, wB1, c1b, cpre);
  k_bn_stats<<<512, 256, 0, stream>>>(cpre, mean, rstd);
  k_bn_apply<<<(8192 * 512 + 255) / 256, 256, 0, stream>>>(cpre, mean, rstd, bn1g, bn1b, bn1o);
  k_conv_gemm<<<cg, 256, 0, stream>>>(bn1o, wB2, c2b, cpre);
  k_bn_stats<<<512, 256, 0, stream>>>(cpre, mean, rstd);
  k_bn_apply<<<(8192 * 512 + 255) / 256, 256, 0, stream>>>(cpre, mean, rstd, bn2g, bn2b, henc);

  k_enc_zero<<<64, 256, 0, stream>>>(hF, cF, hB, cB, hbF, hbB);
  for (int t = 0; t < 256; ++t) {
    k_enc_gates<<<dim3(16, 2), 256, 0, stream>>>(henc, hbF, hbB, whhFb, wihFb,
                                                 whhBb, wihBb, tlens, t, egates);
    k_enc_update<<<dim3(64, 2), 256, 0, stream>>>(egates, bencF, bencB, tlens, t,
                                                  hF, cF, hbF, hB, cB, hbB);
  }

  k_dec_init<<<128, 256, 0, stream>>>(hF, hB, cF, cB, hd, cd, hdb, melb);
  float* out_mel  = (float*)d_out;
  float* out_gate = out_mel + (size_t)32 * 800 * 128;
  float* out_mask = out_gate + (size_t)32 * 800;
  for (int t = 0; t < 800; ++t) {
    k_dec_gates<<<32, 256, 0, stream>>>(hdb, melb, dwhhb, dwihb, dgates);
    k_dec_update_proj<<<1, 256, 0, stream>>>(dgates, bdec, hd, cd, hdb, projwb, projb,
                                             gatew, gateb, mlens, t, melb,
                                             out_mel, out_gate, out_mask);
  }
}
